// WhileOpLSTMNet_9680856285278
// MI455X (gfx1250) — matrix operand order repaired and checked by reading, not compiled, not run on hardware
//
#include <hip/hip_runtime.h>
#include <hip/hip_bf16.h>
#include <stdint.h>

// ---------------------------------------------------------------------------
// 2-layer LSTM, B=64 T=1024 D=H=1024, via v_wmma_f32_16x16x32_bf16.
//  Phase A: convert x and Wg to bf16 (weights stored transposed [N][K]).
//  Phase B (per layer): big GEMM Xpre[r= t*64+b][4096] = A @ WxT + bias
//  Phase C (per layer): 1024 sequential step kernels:
//      gates = Xpre[t] + h_prev @ WhT ; LDS-fused gate math ; update c,h
// Weights (33.5MB bf16) fit in the 192MB L2 -> recurrent re-reads are L2 hits.
// ---------------------------------------------------------------------------

typedef __attribute__((ext_vector_type(8)))  float   v8f;
typedef __attribute__((ext_vector_type(16))) __bf16  v16bf;
typedef __attribute__((ext_vector_type(8)))  __bf16  v8bf;

#define B_   64
#define T_   1024
#define D_   1024
#define HID  1024
#define H4_  4096
#define MTOT (B_ * T_)   // 65536 rows in the big GEMMs

__device__ __forceinline__ __bf16 f2bf(float f) {
  uint32_t u = __builtin_bit_cast(uint32_t, f);
  uint32_t r = (u + 0x7FFFu + ((u >> 16) & 1u)) >> 16;   // round-to-nearest-even
  unsigned short s = (unsigned short)r;
  return __builtin_bit_cast(__bf16, s);
}

__device__ __forceinline__ float sigm(float x) { return 1.0f / (1.0f + __expf(-x)); }

__device__ __forceinline__ v8f wmma_bf16(v16bf a, v16bf b, v8f c) {
  // (neg_a, A, neg_b, B, c_mod, C, reuse_a, reuse_b)
  return __builtin_amdgcn_wmma_f32_16x16x32_bf16(false, a, false, b, (short)0, c,
                                                 false, false);
}

union ABu { v16bf v; v8bf h[2]; };

// ---------------- conversion / init kernels ----------------

__global__ void k_f32_to_bf16(const float* __restrict__ in,
                              __bf16* __restrict__ out, size_t n) {
  size_t i = (size_t)blockIdx.x * blockDim.x + threadIdx.x;
  if (i < n) out[i] = f2bf(in[i]);
}

// wT[mat][n][k] = Wg[l][part*1024 + k][n],  mat = l*2 + part (part0=Wx, part1=Wh)
__global__ void k_wconv(const float* __restrict__ Wg, __bf16* __restrict__ wT) {
  size_t i = (size_t)blockIdx.x * blockDim.x + threadIdx.x;  // < 4*4096*1024
  int k   = (int)(i & 1023);
  int n   = (int)((i >> 10) & 4095);
  int mat = (int)(i >> 22);
  int l = mat >> 1, part = mat & 1;
  float v = Wg[((size_t)(l * 2048 + part * 1024 + k)) * 4096 + n];
  wT[i] = f2bf(v);
}

__global__ void k_zero_u32(uint32_t* __restrict__ p, size_t n) {
  size_t i = (size_t)blockIdx.x * blockDim.x + threadIdx.x;
  if (i < n) p[i] = 0u;
}

// ---------------- big input-projection GEMM ----------------
// xpre[r][n] = sum_k A(row r)[k] * wT[n][k] + bias[n]
// A row r lives at A + (r%64)*strideB + (r/64)*strideT  (handles both the
// [B,T,D] layout of x and the [T,B,H] layout of the layer-0 h history).
// Block: 256 thr / 8 waves. Block covers M-tile (16 rows) x 512 cols.
__global__ void __launch_bounds__(256)
k_gemm_xpre(const __bf16* __restrict__ A, const __bf16* __restrict__ wT,
            const float* __restrict__ bias, float* __restrict__ xpre,
            size_t strideB, size_t strideT) {
  const int lane = threadIdx.x & 31;
  const int wave = threadIdx.x >> 5;
  const int lm   = lane & 15;       // A: row-in-tile, B/C: col-in-tile
  const int hi   = lane >> 4;       // lane-half selects K sub-range
  const int mtile = blockIdx.x;                 // 0..4095
  const int n0    = blockIdx.y * 512 + wave * 64;

  const size_t r = (size_t)mtile * 16 + lm;
  const __bf16* arow = A + (r & 63) * strideB + (r >> 6) * strideT;

  v8f acc[4] = {{}, {}, {}, {}};

  for (int k0 = 0; k0 < 1024; k0 += 32) {
    ABu a;
    const __bf16* ap = arow + k0 + (hi ? 16 : 0);
    a.h[0] = *(const v8bf*)(ap);        // K {0..7} / {16..23}
    a.h[1] = *(const v8bf*)(ap + 8);    // K {8..15} / {24..31}
    const int kb = k0 + (hi ? 16 : 0);
#pragma unroll
    for (int nt = 0; nt < 4; ++nt) {
      const __bf16* bp = wT + (size_t)(n0 + nt * 16 + lm) * 1024 + kb;
      v16bf bfrag = *(const v16bf*)bp;  // 16 contiguous K for this lane's col
      acc[nt] = wmma_bf16(a.v, bfrag, acc[nt]);
    }
  }

#pragma unroll
  for (int nt = 0; nt < 4; ++nt) {
    const int n = n0 + nt * 16 + lm;
    const float bv = bias[n];
#pragma unroll
    for (int e = 0; e < 8; ++e) {
      const size_t row = (size_t)mtile * 16 + hi * 8 + e;  // C layout: M=e / 8+e
      xpre[row * (size_t)H4_ + n] = acc[nt][e] + bv;
    }
  }
}

// ---------------- one LSTM timestep ----------------
// grid 64 blocks x 256 thr. Block jb owns cols j0=jb*16 of every gate.
// Wave w: M-tile (w&3), gate pair (w>>2)*2. Gates staged through LDS so the
// same block fuses the recurrent GEMM with the c/h elementwise update.
__global__ void __launch_bounds__(256)
k_lstm_step(const __bf16* __restrict__ hprev, const __bf16* __restrict__ whT,
            const float* __restrict__ xpre_t, float* __restrict__ cbuf,
            __bf16* __restrict__ hnext, __bf16* __restrict__ hstore_bf,
            float* __restrict__ hstore_f32, int store_f32) {
  __shared__ float gbuf[4][64][16];   // [gate][batch][col] = 16 KB

  const int lane = threadIdx.x & 31;
  const int wave = threadIdx.x >> 5;
  const int lm = lane & 15;
  const int hi = lane >> 4;
  const int mt = wave & 3;            // M-tile: batch rows mt*16..mt*16+15
  const int g0 = (wave >> 2) * 2;     // this wave's first gate (0 or 2)
  const int j0 = blockIdx.x * 16;     // col block within each gate

  const __bf16* arow  = hprev + (size_t)(mt * 16 + lm) * HID;
  const __bf16* brow0 = whT + (size_t)((g0 + 0) * 1024 + j0 + lm) * HID;
  const __bf16* brow1 = whT + (size_t)((g0 + 1) * 1024 + j0 + lm) * HID;

  v8f acc0 = {}, acc1 = {};
  for (int k0 = 0; k0 < 1024; k0 += 32) {
    ABu a;
    const __bf16* ap = arow + k0 + (hi ? 16 : 0);
    a.h[0] = *(const v8bf*)(ap);
    a.h[1] = *(const v8bf*)(ap + 8);
    const int kb = k0 + (hi ? 16 : 0);
    v16bf b0 = *(const v16bf*)(brow0 + kb);
    v16bf b1 = *(const v16bf*)(brow1 + kb);
    acc0 = wmma_bf16(a.v, b0, acc0);
    acc1 = wmma_bf16(a.v, b1, acc1);
  }

  // add precomputed input projection (+bias) and stage gates in LDS
  {
    const int n0 = (g0 + 0) * 1024 + j0 + lm;
    const int n1 = (g0 + 1) * 1024 + j0 + lm;
#pragma unroll
    for (int e = 0; e < 8; ++e) {
      const int brow = mt * 16 + hi * 8 + e;   // batch index
      gbuf[g0 + 0][brow][lm] = acc0[e] + xpre_t[(size_t)brow * H4_ + n0];
      gbuf[g0 + 1][brow][lm] = acc1[e] + xpre_t[(size_t)brow * H4_ + n1];
    }
  }
  __syncthreads();

  // fused gate nonlinearity + state update: 64 batch x 16 cols per block
  for (int idx = threadIdx.x; idx < 64 * 16; idx += 256) {
    const int b  = idx >> 4;
    const int jj = idx & 15;
    const int j  = j0 + jj;
    const float f = gbuf[0][b][jj];
    const float i = gbuf[1][b][jj];
    const float o = gbuf[2][b][jj];
    const float g = gbuf[3][b][jj];
    const size_t ci = (size_t)b * HID + j;
    const float c_new = cbuf[ci] * sigm(f) + tanhf(g) * sigm(i);
    const float h_new = tanhf(c_new) * sigm(o);
    cbuf[ci] = c_new;
    hnext[ci] = f2bf(h_new);
    if (store_f32) hstore_f32[ci] = h_new;      // layer 1 -> final output
    else           hstore_bf[ci]  = f2bf(h_new); // layer 0 -> history for L1 GEMM
  }
}

// ---------------- host orchestration ----------------

extern "C" void kernel_launch(void* const* d_in, const int* in_sizes, int n_in,
                              void* d_out, int out_size, void* d_ws, size_t ws_size,
                              hipStream_t stream) {
  (void)in_sizes; (void)n_in; (void)out_size; (void)ws_size;
  const float* x  = (const float*)d_in[0];   // [64,1024,1024]
  const float* Wg = (const float*)d_in[1];   // [2,2048,4096]
  const float* bg = (const float*)d_in[2];   // [2,4096]
  float* out = (float*)d_out;                // [1024,64,1024] time-major

  char* ws = (char*)d_ws;
  size_t off = 0;
  auto alloc = [&](size_t bytes) -> char* {
    char* p = ws + off;
    off = (off + bytes + 255) & ~(size_t)255;
    return p;
  };
  __bf16* x_bf   = (__bf16*)alloc((size_t)B_ * T_ * D_ * 2);       // 134 MB
  __bf16* wT     = (__bf16*)alloc((size_t)4 * 4096 * 1024 * 2);    //  34 MB
  __bf16* h0_all = (__bf16*)alloc((size_t)MTOT * HID * 2);         // 134 MB
  __bf16* hbuf0  = (__bf16*)alloc((size_t)B_ * HID * 2);
  __bf16* hbuf1  = (__bf16*)alloc((size_t)B_ * HID * 2);
  float*  cbuf   = (float*) alloc((size_t)B_ * HID * 4);
  float*  xpre   = (float*) alloc((size_t)MTOT * H4_ * 4);         // 1 GB (reused per layer)

  // Phase A: bf16 conversions
  {
    size_t n = (size_t)B_ * T_ * D_;
    k_f32_to_bf16<<<dim3((unsigned)((n + 255) / 256)), dim3(256), 0, stream>>>(x, x_bf, n);
  }
  {
    size_t n = (size_t)4 * 4096 * 1024;
    k_wconv<<<dim3((unsigned)(n / 256)), dim3(256), 0, stream>>>(Wg, wT);
  }

  for (int layer = 0; layer < 2; ++layer) {
    // zero h_prev (both buffers) + c state: contiguous 512 KB region
    k_zero_u32<<<dim3(512), dim3(256), 0, stream>>>((uint32_t*)hbuf0, 131072);

    // Phase B: input projection for all timesteps (M=65536, K=1024, N=4096)
    const __bf16* A  = (layer == 0) ? x_bf : h0_all;
    const size_t  sB = (layer == 0) ? (size_t)T_ * D_ : (size_t)HID;       // x:[B,T,D] / h:[T,B,H]
    const size_t  sT = (layer == 0) ? (size_t)D_      : (size_t)B_ * HID;
    const __bf16* wx = wT + (size_t)(layer * 2 + 0) * 4096 * 1024;
    const __bf16* wh = wT + (size_t)(layer * 2 + 1) * 4096 * 1024;
    const float*  bias = bg + (size_t)layer * 4096;
    k_gemm_xpre<<<dim3(MTOT / 16, 8), dim3(256), 0, stream>>>(A, wx, bias, xpre, sB, sT);

    // Phase C: sequential scan over time
    for (int t = 0; t < T_; ++t) {
      const __bf16* hp = (t & 1) ? hbuf1 : hbuf0;
      __bf16*       hn = (t & 1) ? hbuf0 : hbuf1;
      k_lstm_step<<<dim3(64), dim3(256), 0, stream>>>(
          hp, wh, xpre + (size_t)t * B_ * H4_, cbuf, hn,
          h0_all + (size_t)t * B_ * HID,
          out    + (size_t)t * B_ * HID, layer);
    }
  }
}
